// Qwen3Attention_1571958030872
// MI455X (gfx1250) — compile-verified, matrix operand order checked by reading, not run on hardware
//
#include <hip/hip_runtime.h>
#include <hip/hip_bf16.h>
#include <math.h>

// ---------------------------------------------------------------------------
// Types for CDNA5 WMMA (wave32): A/B = v16h (8 VGPR), C/D = v8f (8 VGPR)
// ---------------------------------------------------------------------------
typedef __attribute__((ext_vector_type(16))) _Float16 v16h;
typedef __attribute__((ext_vector_type(8)))  _Float16 v8h;
typedef __attribute__((ext_vector_type(8)))  float    v8f;

#define WMMA_F16(a, b, c) \
  __builtin_amdgcn_wmma_f32_16x16x32_f16(false, (a), false, (b), (short)0, (c), false, false)

// CDNA5 async global->LDS staging (ASYNCcnt path); guarded so the file
// compiles on toolchains that lack the builtins (falls back to sync copies).
#if __has_builtin(__builtin_amdgcn_global_load_async_to_lds_b128) && \
    __has_builtin(__builtin_amdgcn_s_wait_asynccnt)
#define HAVE_ASYNC_LDS 1
#else
#define HAVE_ASYNC_LDS 0
#endif

#if HAVE_ASYNC_LDS
// Builtin prototype is "vV4i*1V4i*3IiIi": (global v4i*, lds v4i*, imm, imm),
// with v4i = GCC-style vector_size(16) int (per hipcc's diagnostic).
typedef int i32x4 __attribute__((vector_size(16)));
typedef __attribute__((address_space(1))) i32x4 as1_i32x4;
typedef __attribute__((address_space(3))) i32x4 as3_i32x4;
#endif

// Copy 16B from global to LDS (per-lane addresses).
__device__ __forceinline__ void cp_b128(const _Float16* g, _Float16* l) {
#if HAVE_ASYNC_LDS
  // Generic shared addr = {aperture[63:32], lds_offset[31:0]}: truncation gives
  // the LDS offset; global generic addresses are identity-mapped.
  __builtin_amdgcn_global_load_async_to_lds_b128(
      (as1_i32x4*)(size_t)(const void*)g,
      (as3_i32x4*)(unsigned)(size_t)(void*)l,
      0, 0);
#else
  *(v8h*)l = *(const v8h*)g;
#endif
}

// Load one 16x32 f16 operand fragment (A layout; B mirrored = same code):
// row = lane&15, K chunks at 8*hi and 16+8*hi. Caller passes pointer already
// offset by row*pitch + 8*hi.
__device__ __forceinline__ v16h ldfrag(const _Float16* p) {
  v8h c0 = *(const v8h*)(p);        // K = 8*hi .. 8*hi+7       -> VGPR 0..3
  v8h c1 = *(const v8h*)(p + 16);   // K = 16+8*hi .. 16+8*hi+7 -> VGPR 4..7
  return __builtin_shufflevector(c0, c1, 0, 1, 2, 3, 4, 5, 6, 7,
                                        8, 9, 10, 11, 12, 13, 14, 15);
}

__device__ __forceinline__ float rmax16(float v) {
  v = fmaxf(v, __shfl_xor(v, 1, 32));
  v = fmaxf(v, __shfl_xor(v, 2, 32));
  v = fmaxf(v, __shfl_xor(v, 4, 32));
  v = fmaxf(v, __shfl_xor(v, 8, 32));
  return v;
}
__device__ __forceinline__ float rsum16(float v) {
  v += __shfl_xor(v, 1, 32);
  v += __shfl_xor(v, 2, 32);
  v += __shfl_xor(v, 4, 32);
  v += __shfl_xor(v, 8, 32);
  return v;
}

// ---------------------------------------------------------------------------
// fp32 -> fp16 conversion
// ---------------------------------------------------------------------------
__global__ void f32_to_f16_kernel(const float* __restrict__ src,
                                  _Float16* __restrict__ dst, int n) {
  int i = blockIdx.x * blockDim.x + threadIdx.x;
  if (i < n) dst[i] = (_Float16)src[i];
}

// ---------------------------------------------------------------------------
// GEMM: C[M,N] = A[M,K] * B[N,K]^T  (A,B f16 row-major K-contiguous, C f32).
// Block = 256 threads = 8 waves (2m x 4n); block tile 64(M) x 256(N);
// wave tile 32x64 = 2x4 WMMA tiles; K stepped by 32 with double-buffered
// async global->LDS staging. M%64==0, N%256==0, K%32==0 (true for all uses).
// LDS pitch 40 halfs: 16B-aligned rows, gcd(20 dw,64 banks)=4 -> low conflict.
// ---------------------------------------------------------------------------
__global__ __launch_bounds__(256) void gemm_f16f32(
    const _Float16* __restrict__ A, const _Float16* __restrict__ B,
    float* __restrict__ C, int M, int N, int K) {
  constexpr int PITCH = 40;
  __shared__ _Float16 As[2][64 * PITCH];    //  5.0 KB per buffer
  __shared__ _Float16 Bs[2][256 * PITCH];   // 20.0 KB per buffer

  const int tid  = threadIdx.x;
  const int lane = tid & 31;
  const int wid  = tid >> 5;
  const int ln = lane & 15, hi = lane >> 4;
  const int wm = wid >> 2, wn = wid & 3;
  const int m0b = blockIdx.y * 64;
  const int n0b = blockIdx.x * 256;

  // Staging work split: thread t covers row t>>2, 16B chunk (t&3).
  const int crow = tid >> 2;        // 0..63
  const int cchk = (tid & 3) * 8;   // half offset 0,8,16,24

  const _Float16* Ag = A + (size_t)(m0b + crow) * K + cchk;
  const _Float16* Bg = B + (size_t)(n0b + crow) * K + cchk;

  auto stage = [&](int ks, int buf) {
    const int k0 = ks * 32;
    cp_b128(Ag + k0, &As[buf][crow * PITCH + cchk]);
#pragma unroll
    for (int i = 0; i < 4; ++i)
      cp_b128(Bg + (size_t)(64 * i) * K + k0,
              &Bs[buf][(crow + 64 * i) * PITCH + cchk]);
  };

  v8f acc[2][4] = {};
  const int nk = K / 32;

  stage(0, 0);
  stage(1, 1);
  for (int kk = 0; kk < nk; ++kk) {
    const int buf = kk & 1;
#if HAVE_ASYNC_LDS
    if (kk + 2 < nk) __builtin_amdgcn_s_wait_asynccnt(5);  // oldest 5 (buf kk) landed
    else             __builtin_amdgcn_s_wait_asynccnt(0);
#endif
    __syncthreads();  // all waves' staged data resident

    const _Float16* Ab = &As[buf][(wm * 32 + ln) * PITCH + 8 * hi];
    const _Float16* Bb = &Bs[buf][(wn * 64 + ln) * PITCH + 8 * hi];
    v16h a0 = ldfrag(Ab);
    v16h a1 = ldfrag(Ab + 16 * PITCH);
#pragma unroll
    for (int j = 0; j < 4; ++j) {
      v16h bj = ldfrag(Bb + j * 16 * PITCH);
      acc[0][j] = WMMA_F16(a0, bj, acc[0][j]);
      acc[1][j] = WMMA_F16(a1, bj, acc[1][j]);
    }
    __syncthreads();  // everyone done reading buf before it is re-staged
    if (kk + 2 < nk) stage(kk + 2, buf);
  }

#pragma unroll
  for (int i = 0; i < 2; ++i) {
#pragma unroll
    for (int r = 0; r < 8; ++r) {  // C/D layout: lane ln = column, VGPR r -> row
      const size_t row = (size_t)(m0b + wm * 32 + i * 16 + r + 8 * hi) * N
                       + n0b + wn * 64 + ln;
#pragma unroll
      for (int j = 0; j < 4; ++j) C[row + j * 16] = acc[i][j][r];
    }
  }
}

// ---------------------------------------------------------------------------
// RoPE + layout change: src f32 [B,S,nh*128] -> dst f16 [B,nh,S,128], *scale.
// ---------------------------------------------------------------------------
__global__ void rope_kernel(const float* __restrict__ src,
                            _Float16* __restrict__ dst,
                            int nheads, float scale, int total) {
  int idx = blockIdx.x * blockDim.x + threadIdx.x;
  if (idx >= total) return;
  int j = idx & 63;
  int t = idx >> 6;
  int h = t % nheads; t /= nheads;
  int s = t % 2048;
  int b = t / 2048;
  size_t row = (size_t)(b * 2048 + s) * ((size_t)nheads * 128) + (size_t)h * 128;
  float x1 = src[row + j];
  float x2 = src[row + j + 64];
  float inv = __expf((float)j * -0.14391156516685312f);  // 10000^(-j/64)
  float th = (float)s * inv;
  float c = cosf(th), sn = sinf(th);
  size_t drow = ((size_t)(b * nheads + h) * 2048 + s) * 128;
  dst[drow + j]      = (_Float16)((x1 * c - x2 * sn) * scale);
  dst[drow + j + 64] = (_Float16)((x2 * c + x1 * sn) * scale);
}

// V: f32 [B,S,4*128] -> f16 transposed [B,4,128,S] (kv position contiguous,
// so PV B-fragments load contiguously).
__global__ void vtrans_kernel(const float* __restrict__ src,
                              _Float16* __restrict__ dst, int total) {
  int idx = blockIdx.x * blockDim.x + threadIdx.x;
  if (idx >= total) return;
  int d = idx & 127;
  int t = idx >> 7;
  int kv = t & 3; t >>= 2;
  int s = t & 2047;
  int b = t >> 11;
  dst[((size_t)(b * 4 + kv) * 128 + d) * 2048 + s] =
      (_Float16)src[(size_t)(b * 2048 + s) * 512 + (size_t)kv * 128 + d];
}

// ---------------------------------------------------------------------------
// Flash attention. Q f16 [B,16,S,128] (pre-scaled by 1/sqrt(128)),
// K f16 [B,4,S,128], Vt f16 [B,4,128,S], mask f32 [B,1,S,S],
// out Ah f16 [B,S,16*128]. Grid: (S/128, B*16); 8 waves x 16 q-rows each.
// ---------------------------------------------------------------------------
__global__ __launch_bounds__(256) void attn_kernel(
    const _Float16* __restrict__ Q, const _Float16* __restrict__ K,
    const _Float16* __restrict__ Vt, const float* __restrict__ mask,
    _Float16* __restrict__ Ah) {
  __shared__ _Float16 pbuf[8 * 512];  // per-wave 16x32 P tile (C->A relayout)

  const int lane = threadIdx.x & 31;
  const int wid  = threadIdx.x >> 5;
  const int ln = lane & 15, hi = lane >> 4;
  const int b   = blockIdx.y >> 4;
  const int h   = blockIdx.y & 15;
  const int kvh = h >> 2;                       // repeat_interleave(4)
  const int q0  = blockIdx.x * 128 + wid * 16;  // this wave's query base

  const _Float16* Qp = Q + ((size_t)(b * 16 + h) * 2048 + q0 + ln) * 128 + 8 * hi;
  v16h qf[4];
#pragma unroll
  for (int kc = 0; kc < 4; ++kc) qf[kc] = ldfrag(Qp + kc * 32);

  const _Float16* Kh = K  + (size_t)(b * 4 + kvh) * 2048 * 128;
  const _Float16* Vh = Vt + (size_t)(b * 4 + kvh) * 128 * 2048;
  const float* mrow0 = mask + ((size_t)b * 2048 + q0) * 2048;

  v8f o[8];
#pragma unroll
  for (int dt = 0; dt < 8; ++dt) o[dt] = (v8f){};
  float mr[8], lr[8];
#pragma unroll
  for (int r = 0; r < 8; ++r) { mr[r] = -1e30f; lr[r] = 0.f; }

  _Float16* pl = pbuf + wid * 512;

  for (int j0 = 0; j0 < 2048; j0 += 32) {
    // ---- S = Q K^T for a 16x32 kv block (2 WMMA n-tiles, 4 k-chunks) ----
    const _Float16* Kp = Kh + (size_t)(j0 + ln) * 128 + 8 * hi;
    v8f s0 = (v8f){}, s1 = (v8f){};
#pragma unroll
    for (int kc = 0; kc < 4; ++kc) {
      v16h b0 = ldfrag(Kp + kc * 32);
      v16h b1 = ldfrag(Kp + (size_t)16 * 128 + kc * 32);
      s0 = WMMA_F16(qf[kc], b0, s0);
      s1 = WMMA_F16(qf[kc], b1, s1);
    }
    // ---- mask add + online softmax (row = r + 8*hi, col = ln / 16+ln) ----
#pragma unroll
    for (int r = 0; r < 8; ++r) {
      const size_t mo = (size_t)(r + 8 * hi) * 2048 + j0 + ln;
      float v0 = s0[r] + mrow0[mo];
      float v1 = s1[r] + mrow0[mo + 16];
      float mx = rmax16(fmaxf(v0, v1));
      float mn = fmaxf(mr[r], mx);
      float alpha = __expf(mr[r] - mn);
      mr[r] = mn;
      float p0 = __expf(v0 - mn);
      float p1 = __expf(v1 - mn);
      pl[(r + 8 * hi) * 32 + ln]      = (_Float16)p0;  // P in q-major layout
      pl[(r + 8 * hi) * 32 + 16 + ln] = (_Float16)p1;
      lr[r] = lr[r] * alpha + rsum16(p0 + p1);
#pragma unroll
      for (int dt = 0; dt < 8; ++dt) o[dt][r] *= alpha;
    }
    // ---- O += P V (A-frag of P from LDS; B-frags contiguous from Vt) ----
    v16h pf = ldfrag(pl + ln * 32 + 8 * hi);
#pragma unroll
    for (int dt = 0; dt < 8; ++dt) {
      v16h vb = ldfrag(Vh + (size_t)(dt * 16 + ln) * 2048 + j0 + 8 * hi);
      o[dt] = WMMA_F16(pf, vb, o[dt]);
    }
  }
  // ---- normalize and write [B,S,16*128] ----
#pragma unroll
  for (int r = 0; r < 8; ++r) {
    float inv = 1.0f / lr[r];
    size_t orow = ((size_t)b * 2048 + q0 + r + 8 * hi) * 2048 + (size_t)h * 128;
#pragma unroll
    for (int dt = 0; dt < 8; ++dt)
      Ah[orow + dt * 16 + ln] = (_Float16)(o[dt][r] * inv);
  }
}

// ---------------------------------------------------------------------------
// Host driver
// ---------------------------------------------------------------------------
extern "C" void kernel_launch(void* const* d_in, const int* in_sizes, int n_in,
                              void* d_out, int out_size, void* d_ws, size_t ws_size,
                              hipStream_t stream) {
  const float* x    = (const float*)d_in[0];
  const float* mask = (const float*)d_in[1];
  const float* wq   = (const float*)d_in[2];
  const float* wk   = (const float*)d_in[3];
  const float* wv   = (const float*)d_in[4];
  const float* wo   = (const float*)d_in[5];
  float* out = (float*)d_out;

  constexpr int B = 2, S = 2048, H = 2048, NH = 16, NKV = 4, HD = 128;
  constexpr int M    = B * S;      // 4096 rows
  constexpr int NQ   = NH * HD;    // 2048
  constexpr int NKVD = NKV * HD;   // 512

  char* ws = (char*)d_ws;
  size_t off = 0;
  auto carve = [&](size_t bytes) -> void* {
    void* p = ws + off;
    off += (bytes + 255) & ~(size_t)255;
    return p;
  };
  _Float16* Xh  = (_Float16*)carve((size_t)M * H * 2);
  _Float16* WqH = (_Float16*)carve((size_t)NQ * H * 2);
  _Float16* WkH = (_Float16*)carve((size_t)NKVD * H * 2);
  _Float16* WvH = (_Float16*)carve((size_t)NKVD * H * 2);
  _Float16* WoH = (_Float16*)carve((size_t)H * H * 2);
  float*    Qf  = (float*)carve((size_t)M * NQ * 4);
  float*    Kf  = (float*)carve((size_t)M * NKVD * 4);
  float*    Vf  = (float*)carve((size_t)M * NKVD * 4);
  _Float16* Qh  = (_Float16*)carve((size_t)M * NQ * 2);
  _Float16* Kh  = (_Float16*)carve((size_t)M * NKVD * 2);
  _Float16* Vt  = (_Float16*)carve((size_t)M * NKVD * 2);
  _Float16* Ah  = (_Float16*)carve((size_t)M * H * 2);
  (void)ws_size; (void)in_sizes; (void)n_in; (void)out_size;

  // fp32 -> fp16 operand conversion
  int n0 = M * H;
  f32_to_f16_kernel<<<(n0 + 255) / 256, 256, 0, stream>>>(x, Xh, n0);
  int n1 = NQ * H;
  f32_to_f16_kernel<<<(n1 + 255) / 256, 256, 0, stream>>>(wq, WqH, n1);
  int n2 = NKVD * H;
  f32_to_f16_kernel<<<(n2 + 255) / 256, 256, 0, stream>>>(wk, WkH, n2);
  f32_to_f16_kernel<<<(n2 + 255) / 256, 256, 0, stream>>>(wv, WvH, n2);
  int n3 = H * H;
  f32_to_f16_kernel<<<(n3 + 255) / 256, 256, 0, stream>>>(wo, WoH, n3);

  // QKV projections (x @ W^T): block tile 64 x 256
  gemm_f16f32<<<dim3(NQ / 256,   M / 64), 256, 0, stream>>>(Xh, WqH, Qf, M, NQ,   H);
  gemm_f16f32<<<dim3(NKVD / 256, M / 64), 256, 0, stream>>>(Xh, WkH, Kf, M, NKVD, H);
  gemm_f16f32<<<dim3(NKVD / 256, M / 64), 256, 0, stream>>>(Xh, WvH, Vf, M, NKVD, H);

  // RoPE (+1/sqrt(hd) folded into Q) and layout transforms
  int tq = B * S * NH * 64;
  rope_kernel<<<(tq + 255) / 256, 256, 0, stream>>>(Qf, Qh, NH, 0.08838834764831845f, tq);
  int tk = B * S * NKV * 64;
  rope_kernel<<<(tk + 255) / 256, 256, 0, stream>>>(Kf, Kh, NKV, 1.0f, tk);
  int tv = B * S * NKV * HD;
  vtrans_kernel<<<(tv + 255) / 256, 256, 0, stream>>>(Vf, Vt, tv);

  // Flash attention
  attn_kernel<<<dim3(S / 128, B * NH), 256, 0, stream>>>(Qh, Kh, Vt, mask, Ah);

  // Output projection -> d_out (f32)
  gemm_f16f32<<<dim3(H / 256, M / 64), 256, 0, stream>>>(Ah, WoH, out, M, H, H);
}
